// WirepointPredictor_22213570855110
// MI455X (gfx1250) — compile-verified, matrix operand order checked by reading, not run on hardware
//
#include <hip/hip_runtime.h>
#include <hip/hip_bf16.h>
#include <math.h>

// ---------------------------------------------------------------------------
// WirepointPredictor for MI455X (gfx1250, wave32, WMMA).
// MLP GEMMs (~70 GFLOP dominant) on v_wmma_f32_16x16x32_bf16, each wave
// register-blocked to a 16x64 tile (4 accumulators, A-fragment reused 4x)
// so the kernel is WMMA-throughput-bound rather than L2-bound.
// ---------------------------------------------------------------------------

typedef __attribute__((ext_vector_type(16))) __bf16 v16bf;
typedef __attribute__((ext_vector_type(8)))  float  v8f;

#define HW      128
#define HW2     16384
#define KTOP    128
#define PPAIR   8128          // 128*127/2
#define BATCH   2
#define NROWS   16256         // BATCH*PPAIR  (divisible by 16)
#define DLOI    128
#define DFC     1024
#define DIN_PAD 1056          // 128*8 + 8 = 1032 padded to 33*32

struct FragU { uint4 lo, hi; };

__device__ __forceinline__ unsigned short f2bf(float x) {
  unsigned u = __float_as_uint(x);
  unsigned r = u + 0x7FFFu + ((u >> 16) & 1u);   // RNE
  return (unsigned short)(r >> 16);
}

// 16-bit A/B fragment load per CDNA5 WMMA layout:
// lanes 0-15 hold K{0..7,16..23}, lanes 16-31 hold K{8..15,24..31} of their row.
__device__ __forceinline__ v16bf load_frag_bf16(const unsigned short* rowbase,
                                                int k0, int lane) {
  int kb = k0 + ((lane >> 4) << 3);
  FragU f;
  f.lo = *reinterpret_cast<const uint4*>(rowbase + kb);
  f.hi = *reinterpret_cast<const uint4*>(rowbase + kb + 16);
  return __builtin_bit_cast(v16bf, f);
}

// ---------------------------------------------------------------------------
// Stage 1: jmap = sigmoid(l1-l0), joff = sigmoid(ch3/4)-0.5
__global__ void jmap_kernel(const float* __restrict__ inputs,
                            float* __restrict__ jmap, float* __restrict__ joff) {
  int t = blockIdx.x * 256 + threadIdx.x;          // BATCH*HW2 threads
  int b = t / HW2, i = t - b * HW2;
  const float* in = inputs + (size_t)b * 5 * HW2;
  float l0 = in[i], l1 = in[HW2 + i];
  jmap[t] = 1.0f / (1.0f + __expf(l0 - l1));
  float c3 = in[3 * HW2 + i], c4 = in[4 * HW2 + i];
  joff[(size_t)b * 2 * HW2 + i]        = 1.0f / (1.0f + __expf(-c3)) - 0.5f;
  joff[(size_t)b * 2 * HW2 + HW2 + i]  = 1.0f / (1.0f + __expf(-c4)) - 0.5f;
}

// 3x3 NMS (SAME, -inf pad == skip OOB)
__global__ void nms_kernel(const float* __restrict__ jmap, float* __restrict__ nmsv) {
  int t = blockIdx.x * 256 + threadIdx.x;
  int b = t / HW2, i = t - b * HW2;
  int h = i / HW, w = i - h * HW;
  const float* jm = jmap + (size_t)b * HW2;
  float v = jm[i], m = v;
  for (int dh = -1; dh <= 1; ++dh) {
    int hh = h + dh; if (hh < 0 || hh >= HW) continue;
    for (int dw = -1; dw <= 1; ++dw) {
      int ww = w + dw; if (ww < 0 || ww >= HW) continue;
      float nv = jm[hh * HW + ww];
      m = nv > m ? nv : m;
    }
  }
  nmsv[t] = (v == m) ? v : 0.0f;
}

// top-k 128, jax tie-break (smaller index wins on ties); then xy coords.
__global__ void topk_kernel(float* __restrict__ nmsv, const float* __restrict__ joff,
                            float* __restrict__ xy, int* __restrict__ selidx) {
  __shared__ unsigned long long skey[256];
  int b = blockIdx.x, tid = threadIdx.x;
  float* vals = nmsv + (size_t)b * HW2;
  for (int iter = 0; iter < KTOP; ++iter) {
    unsigned long long best = 0ull;
    for (int i = tid; i < HW2; i += 256) {
      float v = vals[i];
      if (v >= 0.0f) {
        unsigned long long key =
            (((unsigned long long)__float_as_uint(v)) << 32) | (unsigned)(HW2 - 1 - i);
        if (key > best) best = key;
      }
    }
    skey[tid] = best;
    __syncthreads();
    for (int s = 128; s > 0; s >>= 1) {
      if (tid < s && skey[tid + s] > skey[tid]) skey[tid] = skey[tid + s];
      __syncthreads();
    }
    if (tid == 0) {
      int idx = (HW2 - 1) - (int)(skey[0] & 0xFFFFFFFFull);
      selidx[b * KTOP + iter] = idx;
      vals[idx] = -1.0f;                 // mark taken (regenerated every call)
    }
    __syncthreads();
  }
  if (tid < KTOP) {
    int idx = selidx[b * KTOP + tid];
    float y = (float)(idx / HW) + joff[(size_t)b * 2 * HW2 + idx] + 0.5f;
    float x = (float)(idx % HW) + joff[(size_t)b * 2 * HW2 + HW2 + idx] + 0.5f;
    xy[((size_t)b * KTOP + tid) * 2 + 0] = y;
    xy[((size_t)b * KTOP + tid) * 2 + 1] = x;
  }
}

// triu pair indices
__global__ void pairs_kernel(int* __restrict__ IU, int* __restrict__ IV) {
  int t = blockIdx.x * 256 + threadIdx.x;          // 128*128
  int i = t / KTOP, j = t - i * KTOP;
  if (j > i) {
    int p = i * KTOP - (i * (i + 1)) / 2 + (j - i - 1);
    IU[p] = i; IV[p] = j;
  }
}

// lines output + labels=0 + scores init to b3
__global__ void lines_kernel(const int* __restrict__ IU, const int* __restrict__ IV,
                             const float* __restrict__ xy, const float* __restrict__ b3,
                             float* __restrict__ out) {
  int t = blockIdx.x * 256 + threadIdx.x;
  if (t >= NROWS) return;
  int b = t / PPAIR, p = t - b * PPAIR;
  int u = IU[p], v = IV[p];
  float* lines = out + 2 * NROWS;
  lines[(size_t)t * 4 + 0] = xy[((size_t)b * KTOP + u) * 2 + 0];
  lines[(size_t)t * 4 + 1] = xy[((size_t)b * KTOP + u) * 2 + 1];
  lines[(size_t)t * 4 + 2] = xy[((size_t)b * KTOP + v) * 2 + 0];
  lines[(size_t)t * 4 + 3] = xy[((size_t)b * KTOP + v) * 2 + 1];
  out[t]          = b3[0];   // scores accumulator init
  out[NROWS + t]  = 0.0f;    // labels (Lpos == 0)
}

// ---------------------------------------------------------------------------
// bf16 staging / transposes
__global__ void featT_kernel(const float* __restrict__ features,
                             unsigned short* __restrict__ featT) {
  __shared__ float tile[32][33];
  int bid = blockIdx.x;                 // BATCH * 8 * 512
  int b = bid / (8 * 512);
  int r = bid - b * (8 * 512);
  int c0 = (r / 512) * 32;
  int hw0 = (r % 512) * 32;
  int tx = threadIdx.x & 31, ty = threadIdx.x >> 5;   // ty 0..7
  const float* src = features + (size_t)b * 256 * HW2;
  for (int i = 0; i < 32; i += 8)
    tile[ty + i][tx] = src[(size_t)(c0 + ty + i) * HW2 + hw0 + tx];
  __syncthreads();
  unsigned short* dst = featT + (size_t)b * HW2 * 256;
  for (int i = 0; i < 32; i += 8)
    dst[(size_t)(hw0 + ty + i) * 256 + c0 + tx] = f2bf(tile[tx][ty + i]);
}

__global__ void conv_fc1w_kernel(const float* __restrict__ w, unsigned short* __restrict__ o) {
  int t = blockIdx.x * 256 + threadIdx.x;            // 128*256
  o[t] = f2bf(w[t]);                                  // already (d,c) row-major in K
}

__global__ void conv_w1t_kernel(const float* __restrict__ w1, unsigned short* __restrict__ w1t) {
  int t = blockIdx.x * 256 + threadIdx.x;             // 1024*1056
  int n = t / DIN_PAD, k = t - n * DIN_PAD;
  w1t[t] = (k < 1032) ? f2bf(w1[(size_t)k * DFC + n]) : (unsigned short)0;
}

__global__ void conv_w2t_kernel(const float* __restrict__ w2, unsigned short* __restrict__ w2t) {
  int t = blockIdx.x * 256 + threadIdx.x;             // 1024*1024
  int n = t / DFC, k = t - n * DFC;
  w2t[t] = f2bf(w2[(size_t)k * DFC + n]);
}

// ---------------------------------------------------------------------------
// fc1: xf[b][hw][d] = sum_c featT[b][hw][c] * fc1_w[d][c] + fc1_b[d]
// 16x64 tile per wave: 1 A fragment reused across 4 B fragments per K step.
__global__ void gemm_xf_kernel(const unsigned short* __restrict__ featT,
                               const unsigned short* __restrict__ fc1w,
                               const float* __restrict__ fc1b,
                               float* __restrict__ xf) {
  int wave = threadIdx.x >> 5, lane = threadIdx.x & 31;
  int tile = blockIdx.x * 8 + wave;                   // 4096 wave tiles
  int b = tile >> 11;
  int t = tile & 2047;
  int m0 = (t >> 1) << 4;                             // hw tile
  int n0 = (t & 1) << 6;                              // d macro-tile (64 wide)
  const unsigned short* Arow = featT + ((size_t)b * HW2 + m0 + (lane & 15)) * 256;
  const unsigned short* B0 = fc1w + (size_t)(n0 + (lane & 15)) * 256;
  const unsigned short* B1 = B0 + 16 * 256;
  const unsigned short* B2 = B0 + 32 * 256;
  const unsigned short* B3 = B0 + 48 * 256;
  v8f a0 = {}, a1 = {}, a2 = {}, a3 = {};
  for (int k0 = 0; k0 < 256; k0 += 32) {
    v16bf a = load_frag_bf16(Arow, k0, lane);
    a0 = __builtin_amdgcn_wmma_f32_16x16x32_bf16(false, a, false,
          load_frag_bf16(B0, k0, lane), (short)0, a0, false, false);
    a1 = __builtin_amdgcn_wmma_f32_16x16x32_bf16(false, a, false,
          load_frag_bf16(B1, k0, lane), (short)0, a1, false, false);
    a2 = __builtin_amdgcn_wmma_f32_16x16x32_bf16(false, a, false,
          load_frag_bf16(B2, k0, lane), (short)0, a2, false, false);
    a3 = __builtin_amdgcn_wmma_f32_16x16x32_bf16(false, a, false,
          load_frag_bf16(B3, k0, lane), (short)0, a3, false, false);
  }
  int mrow = m0 + ((lane >> 4) << 3);
  float* out = xf + (size_t)b * HW2 * DLOI;
  v8f accs[4] = {a0, a1, a2, a3};
#pragma unroll
  for (int j = 0; j < 4; ++j) {
    int n = n0 + j * 16 + (lane & 15);
    float bias = fc1b[n];
#pragma unroll
    for (int v = 0; v < 8; ++v)
      out[(size_t)(mrow + v) * DLOI + n] = accs[j][v] + bias;
  }
}

// ---------------------------------------------------------------------------
// LOI pooling -> xv rows (bf16, stride DIN_PAD, tail zero-padded)
__global__ void loi_pool_kernel(const float* __restrict__ xf,
                                const float* __restrict__ out /* lines at +2*NROWS */,
                                unsigned short* __restrict__ xv) {
  __shared__ float sw0[32], sw1[32], sw2[32], sw3[32];
  __shared__ int   so0[32], so1[32], so2[32], so3[32];
  int bp = blockIdx.x;
  int b = bp / PPAIR;
  const float* L = out + 2 * NROWS + (size_t)bp * 4;
  int tid = threadIdx.x;
  if (tid < 32) {
    float yu = L[0], xu = L[1], yv = L[2], xvv = L[3];
    float lam = (float)tid * (1.0f / 31.0f);
    float px = yu * lam + yv * (1.0f - lam) - 0.5f;
    float py = xu * lam + xvv * (1.0f - lam) - 0.5f;
    float px0 = floorf(px); px0 = px0 < 0.f ? 0.f : (px0 > 127.f ? 127.f : px0);
    float py0 = floorf(py); py0 = py0 < 0.f ? 0.f : (py0 > 127.f ? 127.f : py0);
    float px1 = px0 + 1.f; px1 = px1 > 127.f ? 127.f : px1;
    float py1 = py0 + 1.f; py1 = py1 > 127.f ? 127.f : py1;
    sw0[tid] = (px1 - px) * (py1 - py);
    sw1[tid] = (px - px0) * (py1 - py);
    sw2[tid] = (px1 - px) * (py - py0);
    sw3[tid] = (px - px0) * (py - py0);
    so0[tid] = ((int)px0 * HW + (int)py0) * DLOI;
    so1[tid] = ((int)px1 * HW + (int)py0) * DLOI;
    so2[tid] = ((int)px0 * HW + (int)py1) * DLOI;
    so3[tid] = ((int)px1 * HW + (int)py1) * DLOI;
  }
  __syncthreads();
  const float* X = xf + (size_t)b * HW2 * DLOI;
  int d = tid & 127, g = tid >> 7;
  unsigned short* row = xv + (size_t)bp * DIN_PAD;
#pragma unroll
  for (int i = 0; i < 4; ++i) {
    int pt1 = g * 4 + i;
    float best = -3.4e38f;
#pragma unroll
    for (int s = 0; s < 4; ++s) {
      int t = pt1 * 4 + s;
      float v = sw0[t] * X[so0[t] + d] + sw1[t] * X[so1[t] + d] +
                sw2[t] * X[so2[t] + d] + sw3[t] * X[so3[t] + d];
      best = fmaxf(best, v);
    }
    row[d * 8 + pt1] = f2bf(best);      // xv column order: d*8 + pt
  }
  if (tid < 32) row[1024 + tid] = 0;    // feats (==0) + K pad
}

// ---------------------------------------------------------------------------
// h1 = relu(xv @ w1 + b1)   (bf16 out); 16x64 tile per wave.
__global__ void gemm_h1_kernel(const unsigned short* __restrict__ xv,
                               const unsigned short* __restrict__ w1t,
                               const float* __restrict__ b1,
                               unsigned short* __restrict__ h1) {
  int wave = threadIdx.x >> 5, lane = threadIdx.x & 31;
  int tile = blockIdx.x * 8 + wave;                  // 16256 wave tiles
  int m0 = (tile >> 4) << 4;                         // 1016 m-tiles
  int n0 = (tile & 15) << 6;                         // 16 n-macro-tiles (64 wide)
  const unsigned short* Arow = xv  + (size_t)(m0 + (lane & 15)) * DIN_PAD;
  const unsigned short* B0 = w1t + (size_t)(n0 + (lane & 15)) * DIN_PAD;
  const unsigned short* B1 = B0 + (size_t)16 * DIN_PAD;
  const unsigned short* B2 = B0 + (size_t)32 * DIN_PAD;
  const unsigned short* B3 = B0 + (size_t)48 * DIN_PAD;
  v8f a0 = {}, a1 = {}, a2 = {}, a3 = {};
  for (int k0 = 0; k0 < DIN_PAD; k0 += 32) {
    if (k0 + 32 < DIN_PAD) __builtin_prefetch(Arow + k0 + 32, 0, 0);
    v16bf a = load_frag_bf16(Arow, k0, lane);
    a0 = __builtin_amdgcn_wmma_f32_16x16x32_bf16(false, a, false,
          load_frag_bf16(B0, k0, lane), (short)0, a0, false, false);
    a1 = __builtin_amdgcn_wmma_f32_16x16x32_bf16(false, a, false,
          load_frag_bf16(B1, k0, lane), (short)0, a1, false, false);
    a2 = __builtin_amdgcn_wmma_f32_16x16x32_bf16(false, a, false,
          load_frag_bf16(B2, k0, lane), (short)0, a2, false, false);
    a3 = __builtin_amdgcn_wmma_f32_16x16x32_bf16(false, a, false,
          load_frag_bf16(B3, k0, lane), (short)0, a3, false, false);
  }
  int mrow = m0 + ((lane >> 4) << 3);
  v8f accs[4] = {a0, a1, a2, a3};
#pragma unroll
  for (int j = 0; j < 4; ++j) {
    int n = n0 + j * 16 + (lane & 15);
    float bias = b1[n];
#pragma unroll
    for (int v = 0; v < 8; ++v) {
      float t = accs[j][v] + bias;
      h1[(size_t)(mrow + v) * DFC + n] = f2bf(t > 0.f ? t : 0.f);
    }
  }
}

// h2 = relu(h1 @ w2 + b2); scores += h2 @ w3 (fused, shuffle-reduced + atomics)
// 16x64 tile per wave; one atomicAdd per row-half per wave.
__global__ void gemm_h2_score_kernel(const unsigned short* __restrict__ h1,
                                     const unsigned short* __restrict__ w2t,
                                     const float* __restrict__ b2,
                                     const float* __restrict__ w3,
                                     float* __restrict__ scores) {
  int wave = threadIdx.x >> 5, lane = threadIdx.x & 31;
  int tile = blockIdx.x * 8 + wave;                  // 16256 wave tiles
  int m0 = (tile >> 4) << 4;
  int n0 = (tile & 15) << 6;
  const unsigned short* Arow = h1  + (size_t)(m0 + (lane & 15)) * DFC;
  const unsigned short* B0 = w2t + (size_t)(n0 + (lane & 15)) * DFC;
  const unsigned short* B1 = B0 + (size_t)16 * DFC;
  const unsigned short* B2 = B0 + (size_t)32 * DFC;
  const unsigned short* B3 = B0 + (size_t)48 * DFC;
  v8f a0 = {}, a1 = {}, a2 = {}, a3 = {};
  for (int k0 = 0; k0 < DFC; k0 += 32) {
    if (k0 + 32 < DFC) __builtin_prefetch(Arow + k0 + 32, 0, 0);
    v16bf a = load_frag_bf16(Arow, k0, lane);
    a0 = __builtin_amdgcn_wmma_f32_16x16x32_bf16(false, a, false,
          load_frag_bf16(B0, k0, lane), (short)0, a0, false, false);
    a1 = __builtin_amdgcn_wmma_f32_16x16x32_bf16(false, a, false,
          load_frag_bf16(B1, k0, lane), (short)0, a1, false, false);
    a2 = __builtin_amdgcn_wmma_f32_16x16x32_bf16(false, a, false,
          load_frag_bf16(B2, k0, lane), (short)0, a2, false, false);
    a3 = __builtin_amdgcn_wmma_f32_16x16x32_bf16(false, a, false,
          load_frag_bf16(B3, k0, lane), (short)0, a3, false, false);
  }
  int mrow = m0 + ((lane >> 4) << 3);
  v8f accs[4] = {a0, a1, a2, a3};
  float rowsum[8] = {0, 0, 0, 0, 0, 0, 0, 0};
#pragma unroll
  for (int j = 0; j < 4; ++j) {
    int n = n0 + j * 16 + (lane & 15);
    float bias = b2[n];
    float w3n = w3[n];
#pragma unroll
    for (int v = 0; v < 8; ++v) {
      float t = accs[j][v] + bias;
      t = t > 0.f ? t : 0.f;
      rowsum[v] += t * w3n;
    }
  }
#pragma unroll
  for (int v = 0; v < 8; ++v) {
    float s = rowsum[v];
    // reduce over the 16 lanes of this half (same row m, 16 different n)
    s += __shfl_xor(s, 1, 32);
    s += __shfl_xor(s, 2, 32);
    s += __shfl_xor(s, 4, 32);
    s += __shfl_xor(s, 8, 32);
    if ((lane & 15) == 0) atomicAdd(scores + mrow + v, s);
  }
}

// ---------------------------------------------------------------------------
extern "C" void kernel_launch(void* const* d_in, const int* in_sizes, int n_in,
                              void* d_out, int out_size, void* d_ws, size_t ws_size,
                              hipStream_t stream) {
  const float* inputs   = (const float*)d_in[0];
  const float* features = (const float*)d_in[1];
  const float* fc1_w    = (const float*)d_in[2];
  const float* fc1_b    = (const float*)d_in[3];
  const float* w1       = (const float*)d_in[4];
  const float* b1       = (const float*)d_in[5];
  const float* w2       = (const float*)d_in[6];
  const float* b2       = (const float*)d_in[7];
  const float* w3       = (const float*)d_in[8];
  const float* b3       = (const float*)d_in[9];
  float* out = (float*)d_out;   // [scores 16256][labels 16256][lines 65024]

  char* ws = (char*)d_ws;
  size_t o = 0;
  auto take = [&](size_t bytes) { size_t r = o; o += (bytes + 255) & ~(size_t)255; return r; };
  float* jmap  = (float*)(ws + take((size_t)BATCH * HW2 * 4));
  float* nmsv  = (float*)(ws + take((size_t)BATCH * HW2 * 4));
  float* joff  = (float*)(ws + take((size_t)BATCH * 2 * HW2 * 4));
  float* xy    = (float*)(ws + take((size_t)BATCH * KTOP * 2 * 4));
  int*   seli  = (int*)  (ws + take((size_t)BATCH * KTOP * 4));
  int*   IU    = (int*)  (ws + take((size_t)PPAIR * 4));
  int*   IV    = (int*)  (ws + take((size_t)PPAIR * 4));
  unsigned short* featT = (unsigned short*)(ws + take((size_t)BATCH * HW2 * 256 * 2));
  unsigned short* fc1wb = (unsigned short*)(ws + take((size_t)128 * 256 * 2));
  unsigned short* w1t   = (unsigned short*)(ws + take((size_t)DFC * DIN_PAD * 2));
  unsigned short* w2t   = (unsigned short*)(ws + take((size_t)DFC * DFC * 2));
  float*          xf    = (float*)(ws + take((size_t)BATCH * HW2 * DLOI * 4));
  unsigned short* xv    = (unsigned short*)(ws + take((size_t)NROWS * DIN_PAD * 2));
  unsigned short* h1    = (unsigned short*)(ws + take((size_t)NROWS * DFC * 2));

  // line sampling path
  pairs_kernel<<<64, 256, 0, stream>>>(IU, IV);
  jmap_kernel<<<(BATCH * HW2) / 256, 256, 0, stream>>>(inputs, jmap, joff);
  nms_kernel<<<(BATCH * HW2) / 256, 256, 0, stream>>>(jmap, nmsv);
  topk_kernel<<<BATCH, 256, 0, stream>>>(nmsv, joff, xy, seli);
  lines_kernel<<<(NROWS + 255) / 256, 256, 0, stream>>>(IU, IV, xy, b3, out);

  // bf16 staging
  featT_kernel<<<BATCH * 8 * 512, 256, 0, stream>>>(features, featT);
  conv_fc1w_kernel<<<(128 * 256) / 256, 256, 0, stream>>>(fc1_w, fc1wb);
  conv_w1t_kernel<<<(DFC * DIN_PAD) / 256, 256, 0, stream>>>(w1, w1t);
  conv_w2t_kernel<<<(DFC * DFC) / 256, 256, 0, stream>>>(w2, w2t);

  // WMMA GEMMs (16x64 wave tiles)
  gemm_xf_kernel<<<512, 256, 0, stream>>>(featT, fc1wb, fc1_b, xf);
  loi_pool_kernel<<<NROWS, 256, 0, stream>>>(xf, out, xv);
  gemm_h1_kernel<<<2032, 256, 0, stream>>>(xv, w1t, b1, h1);
  gemm_h2_score_kernel<<<2032, 256, 0, stream>>>(h1, w2t, b2, w3, out);
}
